// LFE_81046032876270
// MI455X (gfx1250) — matrix operand order repaired and checked by reading, not compiled, not run on hardware
//
#include <hip/hip_runtime.h>
#include <hip/hip_bf16.h>
#include <hip/hip_fp16.h>
#include <math.h>
#include <stdint.h>

// ---------------------------------------------------------------------------
// MI455X (gfx1250) implementation notes
//
// Roofline: ~130 GFLOP total, ~120 GFLOP of it in the conv1x1 GEMMs
// (W[OxI] x X[Ix4096] per batch, O,I all multiples of 32). ~900 MB of fp32
// traffic -> ~40us at 23.3 TB/s. FP32 VALU would make this compute bound, so
// the GEMMs convert fp32 -> f16 during staging and run on
// v_wmma_f32_16x16x32_f16 (f32 accumulate), keeping the kernel memory-bound.
//
// CDNA5-specific paths used:
//  * v_wmma_f32_16x16x32_f16 for all conv1x1 GEMMs
//  * global_load_async_to_lds_b128 (+ s_wait_asynccnt) for the A/weight tile:
//    256 threads x 16B = the whole 32x32 f32 tile in one async op per thread,
//    overlapping with the B-tile cvt/ds staging (ASYNCcnt-tracked DMA).
//  * global_prefetch_b8 for the next B k-tile.
//  * B staging packs 4 K-values per ds_store_b64 (was 32x ds_store_b16).
// ---------------------------------------------------------------------------

typedef __attribute__((ext_vector_type(16))) _Float16 v16h;
typedef __attribute__((ext_vector_type(8)))  _Float16 v8h;
typedef __attribute__((ext_vector_type(4)))  _Float16 v4h;
typedef __attribute__((ext_vector_type(8)))  float    v8f;

#define BM 32
#define BN 128
#define BK 32

// Out[b][o][n] = sum_i W[o][i] * X[b][i][n]
// Requires O % 32 == 0, I % 32 == 0, NTOT % 128 == 0 (true for all call sites).
__global__ __launch_bounds__(256) void gemm_wmma(
    const float* __restrict__ W, const float* __restrict__ X,
    float* __restrict__ Out, int O, int I, int NTOT)
{
  __shared__ __align__(16) float    As32[BM][BK];  // weights tile, fp32 (async DMA target)
  __shared__ __align__(32) _Float16 Bs[BN][BK];    // [n-col][k] f16, transposed for B frag

  const int b     = blockIdx.z;
  const int oBase = blockIdx.y * BM;
  const int nBase = blockIdx.x * BN;
  const float* __restrict__ Xb = X + (size_t)b * I * NTOT;
  float* __restrict__ Ob       = Out + (size_t)b * O * NTOT;

  const int t    = threadIdx.x;
  const int lane = t & 31;
  const int wave = t >> 5;      // 0..7
  const int wm   = wave & 1;    // M subtile (16 rows each)
  const int wn   = wave >> 1;   // 0..3, 32 cols each

  v8f acc[2] = {};

  // Per-thread staging coordinates
  const int arow = t >> 3;            // 0..31 : A rows, 4 floats (16B) per thread
  const int acol = (t & 7) * 4;
  const int bn1  = t & 127;           // B: thread owns column n, 16 consecutive k
  const int bkh  = (t >> 7) * 16;     //    k-half: 0 or 16

  // LDS byte address of this thread's async-DMA destination (flat -> lds trunc)
  const uint32_t ldsA = (uint32_t)(uintptr_t)&As32[arow][acol];

  for (int k0 = 0; k0 < I; k0 += BK) {
    // ---- stage A tile (weights) via async global->LDS DMA (fp32, 16B/lane) ----
    {
      const float* gsrc = W + (size_t)(oBase + arow) * I + k0 + acol;
      asm volatile("global_load_async_to_lds_b128 %0, %1, off"
                   :: "v"(ldsA), "v"(gsrc) : "memory");
    }
    // ---- stage B tile (activations), transposed, fp32 -> f16, 8B LDS stores ----
    {
      const float* src = Xb + (size_t)(k0 + bkh) * NTOT + nBase + bn1;
      if (k0 + BK < I) __builtin_prefetch(src + (size_t)BK * NTOT, 0, 1);
      #pragma unroll
      for (int g = 0; g < 4; ++g) {
        const float f0 = src[(size_t)(4 * g + 0) * NTOT];
        const float f1 = src[(size_t)(4 * g + 1) * NTOT];
        const float f2 = src[(size_t)(4 * g + 2) * NTOT];
        const float f3 = src[(size_t)(4 * g + 3) * NTOT];
        v4h pk = {(_Float16)f0, (_Float16)f1, (_Float16)f2, (_Float16)f3};
        *reinterpret_cast<v4h*>(&Bs[bn1][bkh + 4 * g]) = pk;   // ds_store_b64
      }
    }
    // async A must land before anyone reads it after the barrier
    asm volatile("s_wait_asynccnt 0x0" ::: "memory");
    __syncthreads();

    // ---- A fragment (from fp32 LDS): lane L -> M = L%16 ; contiguous K
    // to match B: elem j -> K=hs*16+j.
    const int hs = lane >> 4;
    v16h a;
    {
      const int r = wm * 16 + (lane & 15);
      v8f a0 = *reinterpret_cast<const v8f*>(&As32[r][hs * 16]);        // 2x ds_load_b128
      v8f a1 = *reinterpret_cast<const v8f*>(&As32[r][hs * 16 + 8]);
      #pragma unroll
      for (int j = 0; j < 8; ++j) { a[j] = (_Float16)a0[j]; a[8 + j] = (_Float16)a1[j]; }
    }
    // ---- B fragments: lane L -> N = L%16 ; elem j -> K = hs*16 + j ----
    #pragma unroll
    for (int nt = 0; nt < 2; ++nt) {
      const int ncol = wn * 32 + nt * 16 + (lane & 15);
      v16h bf = *reinterpret_cast<const v16h*>(&Bs[ncol][hs * 16]);
      acc[nt] = __builtin_amdgcn_wmma_f32_16x16x32_f16(
          /*neg_a=*/false, a, /*neg_b=*/false, bf,
          /*c_mod=*/(short)0, acc[nt], /*reuse_a=*/false, /*reuse_b=*/false);
    }
    __syncthreads();
  }

  // ---- store D: VGPR r, lane L -> M = r + 8*(L/16), N = L%16 ----
  const int hs = lane >> 4;
  #pragma unroll
  for (int nt = 0; nt < 2; ++nt) {
    const int n = nBase + wn * 32 + nt * 16 + (lane & 15);
    #pragma unroll
    for (int r = 0; r < 8; ++r) {
      const int o = oBase + wm * 16 + r + 8 * hs;
      Ob[(size_t)o * NTOT + n] = acc[nt][r];
    }
  }
}

// LayerNorm over channel dim of [B, C, HW]; one thread per (b, pos).
__global__ void ln_kernel(const float* __restrict__ x, const float* __restrict__ w,
                          const float* __restrict__ bias, float* __restrict__ y,
                          int C, int HW)
{
  const int b   = blockIdx.y;
  const int pos = blockIdx.x * blockDim.x + threadIdx.x;
  if (pos >= HW) return;
  const float* xp = x + (size_t)b * C * HW + pos;
  float mu = 0.f;
  for (int c = 0; c < C; ++c) mu += xp[(size_t)c * HW];
  mu /= (float)C;
  float var = 0.f;
  for (int c = 0; c < C; ++c) { float d = xp[(size_t)c * HW] - mu; var += d * d; }
  var /= (float)C;
  const float inv = rsqrtf(var + 1e-5f);
  float* yp = y + (size_t)b * C * HW + pos;
  for (int c = 0; c < C; ++c)
    yp[(size_t)c * HW] = (xp[(size_t)c * HW] - mu) * inv * w[c] + bias[c];
}

// Depthwise 3x3, zero padded. x,y: [B*Ch, H*W]; wt: [Ch,1,3,3].
__global__ void dw3_kernel(const float* __restrict__ x, const float* __restrict__ wt,
                           float* __restrict__ y, int Ch, int H, int W)
{
  const int bc  = blockIdx.y;      // b*Ch + ch
  const int ch  = bc % Ch;
  const int pos = blockIdx.x * blockDim.x + threadIdx.x;
  if (pos >= H * W) return;
  const int hy = pos / W, wx = pos % W;
  const float* xp = x + (size_t)bc * H * W;
  const float* wp = wt + (size_t)ch * 9;
  float acc = 0.f;
  #pragma unroll
  for (int dy = -1; dy <= 1; ++dy) {
    const int yy = hy + dy;
    if (yy < 0 || yy >= H) continue;
    #pragma unroll
    for (int dx = -1; dx <= 1; ++dx) {
      const int xx = wx + dx;
      if (xx < 0 || xx >= W) continue;
      acc += xp[yy * W + xx] * wp[(dy + 1) * 3 + (dx + 1)];
    }
  }
  y[(size_t)bc * H * W + pos] = acc;
}

// invn[r] = 1 / max(||row||_2, 1e-12); one wave32 per row.
__global__ void rownorm_kernel(const float* __restrict__ base, float* __restrict__ invn,
                               int rowsPerB, int n, int strideCh, int chOff, int totalRows)
{
  const int gw   = (blockIdx.x * blockDim.x + threadIdx.x) >> 5;
  const int lane = threadIdx.x & 31;
  if (gw >= totalRows) return;
  const int b = gw / rowsPerB, ch = gw % rowsPerB;
  const float* p = base + ((size_t)b * strideCh + chOff + ch) * (size_t)n;
  float s = 0.f;
  for (int i = lane; i < n; i += 32) { float v = p[i]; s += v * v; }
  #pragma unroll
  for (int off = 16; off; off >>= 1) s += __shfl_down(s, off, 32);
  if (lane == 0) invn[gw] = 1.0f / fmaxf(sqrtf(s), 1e-12f);
}

// S[bh][ci][dj] = <q_row, k_row> * invq * invk ; one wave32 per (ci,dj) pair.
__global__ void qk_kernel(const float* __restrict__ q, const float* __restrict__ k,
                          const float* __restrict__ invq, const float* __restrict__ invk,
                          float* __restrict__ S, int heads, int cc, int n,
                          int qStride, int kStride)
{
  const int bh = blockIdx.x;
  const int b = bh / heads, h = bh % heads;
  const int wv   = blockIdx.y * (blockDim.x >> 5) + (threadIdx.x >> 5);
  const int lane = threadIdx.x & 31;
  if (wv >= cc * cc) return;
  const int ci = wv / cc, dj = wv % cc;
  const int C = heads * cc;
  const float* qp = q + ((size_t)b * qStride + h * cc + ci) * (size_t)n;
  const float* kp = k + ((size_t)b * kStride + h * cc + dj) * (size_t)n;
  float s = 0.f;
  for (int i = lane; i < n; i += 32) s += qp[i] * kp[i];
  #pragma unroll
  for (int off = 16; off; off >>= 1) s += __shfl_down(s, off, 32);
  if (lane == 0)
    S[((size_t)bh * cc + ci) * cc + dj] =
        s * invq[b * C + h * cc + ci] * invk[b * C + h * cc + dj];
}

// softmax over last dim (cc) with per-head temperature; one thread per row.
__global__ void softmax_kernel(float* __restrict__ S, const float* __restrict__ temp,
                               int heads, int cc, int rows)
{
  const int r = blockIdx.x * blockDim.x + threadIdx.x;
  if (r >= rows) return;
  const int bh = r / cc;
  const float tp = temp[bh % heads];
  float* row = S + (size_t)r * cc;
  float m = -1e30f;
  for (int i = 0; i < cc; ++i) m = fmaxf(m, row[i] * tp);
  float s = 0.f;
  for (int i = 0; i < cc; ++i) { float e = expf(row[i] * tp - m); s += e; row[i] = e; }
  const float inv = 1.0f / s;
  for (int i = 0; i < cc; ++i) row[i] *= inv;
}

#define MAXCC 36
// out[c][n] = sum_d attn[c][d] * v[d][n]; attn + v-tile cached in LDS.
__global__ __launch_bounds__(128) void av_kernel(
    const float* __restrict__ S, const float* __restrict__ vbase,
    float* __restrict__ out, int heads, int cc, int n,
    int vStride, int vOff, int oStride)
{
  __shared__ float Sl[MAXCC * MAXCC];
  __shared__ float Vl[MAXCC][128];
  const int b = blockIdx.z, h = blockIdx.y;
  const int nBase = blockIdx.x * 128;
  const int t = threadIdx.x;
  const float* Sp = S + (size_t)(b * heads + h) * cc * cc;
  for (int i = t; i < cc * cc; i += 128) Sl[i] = Sp[i];
  for (int d = 0; d < cc; ++d)
    Vl[d][t] = vbase[((size_t)b * vStride + vOff + h * cc + d) * (size_t)n + nBase + t];
  __syncthreads();
  for (int c = 0; c < cc; ++c) {
    float acc = 0.f;
    for (int d = 0; d < cc; ++d) acc += Sl[c * cc + d] * Vl[d][t];
    out[((size_t)b * oStride + h * cc + c) * (size_t)n + nBase + t] = acc;
  }
}

// y = gelu_exact(x[:, :C2]) * x[:, C2:2*C2]  with x: [B, 2*C2, HW]
__global__ void gate_kernel(const float* __restrict__ x, float* __restrict__ y,
                            int C2, int HW)
{
  const int bj = blockIdx.y;            // b*C2 + j
  const int b = bj / C2, j = bj % C2;
  const int pos = blockIdx.x * blockDim.x + threadIdx.x;
  if (pos >= HW) return;
  const float v1 = x[((size_t)b * 2 * C2 + j) * HW + pos];
  const float v2 = x[((size_t)b * 2 * C2 + C2 + j) * HW + pos];
  const float g  = 0.5f * v1 * (1.0f + erff(v1 * 0.70710678118654752f));
  y[((size_t)b * C2 + j) * HW + pos] = g * v2;
}

__global__ void add_kernel(const float* __restrict__ a, const float* __restrict__ b,
                           float* __restrict__ y, size_t ntot)
{
  size_t i = (size_t)blockIdx.x * blockDim.x + threadIdx.x;
  const size_t stride = (size_t)gridDim.x * blockDim.x;
  for (; i < ntot; i += stride) y[i] = a[i] + b[i];
}

// ---------------------------------------------------------------------------
// Host-side pipeline for one path (C = 96 low / 288 high).
// ---------------------------------------------------------------------------
static void run_path(
    const float* qin, const float* fin,
    const float* nw1, const float* nb1, const float* nw2, const float* nb2,
    const float* temp,
    const float* aq, const float* aqdw, const float* akv, const float* akvdw,
    const float* apo,
    const float* fw_in, const float* fw_dw, const float* fw_out,
    float* out, int C,
    float* A, float* Q, float* QD, float* KV, float* KVD, float* X1, float* F2,
    float* Sbuf, float* qn, float* kn,
    hipStream_t stream)
{
  const int B = 16, H = 64, Wd = 64, HW = H * Wd, HEADS = 8;
  const int cc = C / HEADS;
  const size_t uC = (size_t)B * C * HW;
  dim3 b256(256);

  // ---- attention half ----
  ln_kernel<<<dim3(HW / 256, B), b256, 0, stream>>>(qin, nw1, nb1, A, C, HW);
  gemm_wmma<<<dim3(HW / BN, C / BM, B), b256, 0, stream>>>(aq, A, Q, C, C, HW);
  dw3_kernel<<<dim3(HW / 256, B * C), b256, 0, stream>>>(Q, aqdw, QD, C, H, Wd);
  gemm_wmma<<<dim3(HW / BN, (2 * C) / BM, B), b256, 0, stream>>>(akv, fin, KV, 2 * C, C, HW);
  dw3_kernel<<<dim3(HW / 256, B * 2 * C), b256, 0, stream>>>(KV, akvdw, KVD, 2 * C, H, Wd);

  const int rows = B * C;
  rownorm_kernel<<<dim3((rows + 7) / 8), b256, 0, stream>>>(QD,  qn, C, HW, C,     0, rows);
  rownorm_kernel<<<dim3((rows + 7) / 8), b256, 0, stream>>>(KVD, kn, C, HW, 2 * C, 0, rows);
  qk_kernel<<<dim3(B * HEADS, (cc * cc + 7) / 8), b256, 0, stream>>>(
      QD, KVD, qn, kn, Sbuf, HEADS, cc, HW, C, 2 * C);
  {
    const int srows = B * HEADS * cc;
    softmax_kernel<<<dim3((srows + 255) / 256), b256, 0, stream>>>(
        Sbuf, temp, HEADS, cc, srows);
  }
  av_kernel<<<dim3(HW / 128, HEADS, B), dim3(128), 0, stream>>>(
      Sbuf, KVD, A, HEADS, cc, HW, 2 * C, C, C);
  gemm_wmma<<<dim3(HW / BN, C / BM, B), b256, 0, stream>>>(apo, A, Q, C, C, HW);
  add_kernel<<<dim3((unsigned)((uC + 255) / 256)), b256, 0, stream>>>(qin, Q, X1, uC);

  // ---- FFN half ----  (F1 aliases KV region: 4*uC floats, KV/KVD are dead)
  float* F1 = KV;
  float* G  = KV;  // gate output reuses the same region after dw reads F1
  ln_kernel<<<dim3(HW / 256, B), b256, 0, stream>>>(X1, nw2, nb2, A, C, HW);
  gemm_wmma<<<dim3(HW / BN, (4 * C) / BM, B), b256, 0, stream>>>(fw_in, A, F1, 4 * C, C, HW);
  dw3_kernel<<<dim3(HW / 256, B * 4 * C), b256, 0, stream>>>(F1, fw_dw, F2, 4 * C, H, Wd);
  gate_kernel<<<dim3(HW / 256, B * 2 * C), b256, 0, stream>>>(F2, G, 2 * C, HW);
  gemm_wmma<<<dim3(HW / BN, C / BM, B), b256, 0, stream>>>(fw_out, G, A, C, 2 * C, HW);
  add_kernel<<<dim3((unsigned)((uC + 255) / 256)), b256, 0, stream>>>(X1, A, out, uC);
}

extern "C" void kernel_launch(void* const* d_in, const int* in_sizes, int n_in,
                              void* d_out, int out_size, void* d_ws, size_t ws_size,
                              hipStream_t stream)
{
  (void)in_sizes; (void)n_in; (void)out_size; (void)ws_size;
  const int B = 16, HW = 64 * 64;

  const float* q_L     = (const float*)d_in[0];
  const float* q_H     = (const float*)d_in[1];   // [B, 96*3, HW] flat
  const float* fuse_L  = (const float*)d_in[2];
  const float* fuse_H  = (const float*)d_in[3];
  const float* n1_w = (const float*)d_in[4],  *n1_b = (const float*)d_in[5];
  const float* n2_w = (const float*)d_in[6],  *n2_b = (const float*)d_in[7];
  const float* n3_w = (const float*)d_in[8],  *n3_b = (const float*)d_in[9];
  const float* n4_w = (const float*)d_in[10], *n4_b = (const float*)d_in[11];
  const float* a1_temp = (const float*)d_in[12];
  const float* a1_q    = (const float*)d_in[13];
  const float* a1_qdw  = (const float*)d_in[14];
  const float* a1_kv   = (const float*)d_in[15];
  const float* a1_kvdw = (const float*)d_in[16];
  const float* a1_po   = (const float*)d_in[17];
  const float* f1_in   = (const float*)d_in[18];
  const float* f1_dw   = (const float*)d_in[19];
  const float* f1_out  = (const float*)d_in[20];
  const float* a2_temp = (const float*)d_in[21];
  const float* a2_q    = (const float*)d_in[22];
  const float* a2_qdw  = (const float*)d_in[23];
  const float* a2_kv   = (const float*)d_in[24];
  const float* a2_kvdw = (const float*)d_in[25];
  const float* a2_po   = (const float*)d_in[26];
  const float* f2_in   = (const float*)d_in[27];
  const float* f2_dw   = (const float*)d_in[28];
  const float* f2_out  = (const float*)d_in[29];

  float* out_x = (float*)d_out;                       // [B, 96, HW]
  float* out_y = out_x + (size_t)B * 96 * HW;         // [B, 288, HW]

  // Workspace layout (floats), sized for the larger path (D = 288):
  //   u = B*288*HW = 18,874,368 ; total ~12u + small  (~907 MB)
  float* ws = (float*)d_ws;
  const size_t u = (size_t)B * 288 * HW;
  float* A   = ws;              // 1u : LN out / attn out / ffn out (reused)
  float* Q   = A + u;           // 1u : q conv out / proj out
  float* QD  = Q + u;           // 1u : q after dwconv
  float* KV  = QD + u;          // 2u : kv conv out ; aliases FFN hidden F1 (4u span)
  float* KVD = KV + 2 * u;      // 2u : k,v after dwconv
  float* X1  = KVD + 2 * u;     // 1u : attention residual
  float* F2  = X1 + u;          // 4u : FFN hidden after dwconv
  float* Sb  = F2 + 4 * u;                      // B*8*36*36 attention matrices
  float* qn  = Sb + (size_t)16 * 8 * 36 * 36;   // B*288 inverse norms
  float* kn  = qn + (size_t)16 * 288;

  // Low-frequency path (C = 96)
  run_path(q_L, fuse_L, n1_w, n1_b, n2_w, n2_b, a1_temp,
           a1_q, a1_qdw, a1_kv, a1_kvdw, a1_po,
           f1_in, f1_dw, f1_out,
           out_x, 96,
           A, Q, QD, KV, KVD, X1, F2, Sb, qn, kn, stream);

  // High-frequency path (D = 288; reshape (B,C,N,H,W)->(B,C*N,H,W) is identity on memory)
  run_path(q_H, fuse_H, n3_w, n3_b, n4_w, n4_b, a2_temp,
           a2_q, a2_qdw, a2_kv, a2_kvdw, a2_po,
           f2_in, f2_dw, f2_out,
           out_y, 288,
           A, Q, QD, KV, KVD, X1, F2, Sb, qn, kn, stream);
}
